// SimpleMoleculeGCN_24790551232515
// MI455X (gfx1250) — matrix operand order repaired and checked by reading, not compiled, not run on hardware
//
#include <hip/hip_runtime.h>

// Problem constants (match reference)
#define N_NODES  100000
#define N_EDGES  1600000
#define F_INx    128
#define H_DIM    128
#define G_GRAPHS 2048
#define BN_EPS   1e-5f

typedef __attribute__((ext_vector_type(16))) _Float16 v16h;
typedef __attribute__((ext_vector_type(8)))  _Float16 v8h;
typedef __attribute__((ext_vector_type(8)))  float    v8f;

// ---------------------------------------------------------------- utilities
__global__ void zero_kernel(float* __restrict__ p, int n) {
  int i = blockIdx.x * blockDim.x + threadIdx.x;
  if (i < n) p[i] = 0.f;
}

// x (f32) -> xh (f16), used once for the layer-1 input
__global__ void f32_to_f16(const float* __restrict__ in,
                           _Float16* __restrict__ out, int n) {
  int i = blockIdx.x * blockDim.x + threadIdx.x;
  if (i < n) out[i] = (_Float16)in[i];
}

// W[k][c] f32 -> Wt[c][k] f16 (pre-transpose so B fragments are contiguous)
__global__ void wt_prep(const float* __restrict__ W, _Float16* __restrict__ Wt) {
  int i = blockIdx.x * blockDim.x + threadIdx.x;  // 16384
  int c = i >> 7;
  int k = i & 127;
  Wt[c * F_INx + k] = (_Float16)W[k * H_DIM + c];
}

// ---------------------------------------------------------------- degrees
__global__ void deg_accum(const int* __restrict__ dst, float* __restrict__ deg) {
  int e = blockIdx.x * blockDim.x + threadIdx.x;
  if (e < N_EDGES) atomicAdd(&deg[dst[e]], 1.f);
}

__global__ void deg_finalize(float* __restrict__ dis) {
  int i = blockIdx.x * blockDim.x + threadIdx.x;
  if (i < N_NODES) dis[i] = rsqrtf(dis[i] + 1.f);  // deg + self-loop
}

// ---------------------------------------------------------------- WMMA GEMM
// Y[N,128] = Xh[N,128] @ W[128,128] + bias, Xh/Wt in f16, acc in f32.
// One block = 16 rows staged in LDS (f16, padded row); 8 waves own the 8
// 16-column tiles; K swept in 4 unrolled chunks of 32 via
// v_wmma_f32_16x16x32_f16.  Fragment packing (contiguous K order chosen here, same for A and B):
//   A 16x32 f16 : lane&15=M; half 0 K=0..15, half 1 K=16..31
//                 -> one contiguous 16-f16 run -> 2x ds_load_b128
//   B 32x16 f16 : lane&15=N; half 0 K=0..15, half 1 K=16..31
//                 -> one contiguous 16-f16 run of Wt[col][*]
//   C/D 16x16 f32: VGPR r -> M = r + 8*half, N = lane&15
__global__ __launch_bounds__(256) void gemm_wmma(
    const _Float16* __restrict__ Xh, const _Float16* __restrict__ Wt,
    const float* __restrict__ bias, float* __restrict__ Y) {
  __shared__ _Float16 As[16][F_INx + 8];

  const int t    = threadIdx.x;
  const int row0 = blockIdx.x * 16;

  // cooperative stage: 16x128 f16 tile, one b128 per thread
  {
    int r = t >> 4;
    int c = (t & 15) * 8;
    *(v8h*)&As[r][c] = *(const v8h*)&Xh[(row0 + r) * F_INx + c];
  }
  __syncthreads();

  const int wave = t >> 5;
  const int lane = t & 31;
  const int half = lane >> 4;
  const int l15  = lane & 15;
  const int col  = wave * 16 + l15;

  v8f acc;
  {
    float bv = bias[col];
    for (int r = 0; r < 8; ++r) acc[r] = bv;
  }

#pragma unroll
  for (int kc = 0; kc < F_INx / 32; ++kc) {
    const int kbase = kc * 32;
    v8h a0 = *(const v8h*)&As[l15][kbase + 16 * half];
    v8h a1 = *(const v8h*)&As[l15][kbase + 16 * half + 8];
    v16h a = __builtin_shufflevector(a0, a1, 0, 1, 2, 3, 4, 5, 6, 7,
                                     8, 9, 10, 11, 12, 13, 14, 15);
    v16h b = *(const v16h*)&Wt[col * F_INx + kbase + 16 * half];
    acc = __builtin_amdgcn_wmma_f32_16x16x32_f16(
        /*neg_a=*/false, a, /*neg_b=*/false, b,
        /*c_mod=*/(short)0, acc, /*reuse_a=*/false, /*reuse_b=*/false);
  }

  for (int r = 0; r < 8; ++r)
    Y[(row0 + r + half * 8) * H_DIM + col] = acc[r];
}

// ---------------------------------------------------------------- GCN scatter
// agg = xw * dis^2 (self loop) — doubles as the accumulator init.
__global__ void selfloop_init(const float* __restrict__ xw,
                              const float* __restrict__ dis,
                              float* __restrict__ agg) {
  int i = blockIdx.x * blockDim.x + threadIdx.x;
  if (i < N_NODES * H_DIM) {
    int node = i >> 7;
    float d = dis[node];
    agg[i] = xw[i] * d * d;
  }
}

// One wave per edge; lane owns 4 consecutive features (b128 gather + 4 atomics)
__global__ void scatter_edges(const float* __restrict__ xw,
                              const int* __restrict__ src,
                              const int* __restrict__ dst,
                              const float* __restrict__ dis,
                              float* __restrict__ agg) {
  int gt   = blockIdx.x * blockDim.x + threadIdx.x;
  int e    = gt >> 5;
  int lane = gt & 31;
  if (e >= N_EDGES) return;
  int s = src[e];
  int d = dst[e];
  float nrm = dis[s] * dis[d];
  const float4 v = *(const float4*)(xw + s * H_DIM + lane * 4);
  float* ad = agg + d * H_DIM + lane * 4;
  atomicAdd(&ad[0], v.x * nrm);
  atomicAdd(&ad[1], v.y * nrm);
  atomicAdd(&ad[2], v.z * nrm);
  atomicAdd(&ad[3], v.w * nrm);
}

// ---------------------------------------------------------------- BatchNorm
// stats: [0..127]=sum  [128..255]=sumsq  [256..383]=scale  [384..511]=shift
__global__ void bn_stats(const float* __restrict__ x, float* __restrict__ stats) {
  int f  = threadIdx.x;  // 128 threads = 128 features
  int r0 = blockIdx.x * 256;
  float s = 0.f, ss = 0.f;
  for (int r = 0; r < 256; ++r) {
    int row = r0 + r;
    if (row < N_NODES) {
      float v = x[row * H_DIM + f];
      s  += v;
      ss += v * v;
    }
  }
  atomicAdd(&stats[f], s);
  atomicAdd(&stats[H_DIM + f], ss);
}

__global__ void bn_finalize(float* __restrict__ stats,
                            const float* __restrict__ g,
                            const float* __restrict__ bt) {
  int f = threadIdx.x;
  float mu  = stats[f] * (1.f / (float)N_NODES);
  float var = stats[H_DIM + f] * (1.f / (float)N_NODES) - mu * mu;
  float inv = rsqrtf(var + BN_EPS);
  float sc  = g[f] * inv;
  stats[2 * H_DIM + f] = sc;
  stats[3 * H_DIM + f] = bt[f] - mu * sc;
}

// BN + ReLU, emitting f16 activations for the next layer's WMMA GEMM
__global__ void bn_apply_relu(const float* __restrict__ in,
                              const float* __restrict__ stats,
                              _Float16* __restrict__ out) {
  int i = blockIdx.x * blockDim.x + threadIdx.x;
  if (i < N_NODES * H_DIM) {
    int f = i & 127;
    float v = in[i] * stats[2 * H_DIM + f] + stats[3 * H_DIM + f];
    out[i] = (_Float16)(v > 0.f ? v : 0.f);
  }
}

// ---------------------------------------------------------------- pooling + head
__global__ void pool_accum(const _Float16* __restrict__ h,
                           const int* __restrict__ batch,
                           float* __restrict__ sums, float* __restrict__ cnts) {
  int node = blockIdx.x;
  int f    = threadIdx.x;
  int g    = batch[node];
  atomicAdd(&sums[g * H_DIM + f], (float)h[node * H_DIM + f]);
  if (f == 0) atomicAdd(&cnts[g], 1.f);
}

__global__ __launch_bounds__(64) void fc_head(
    const float* __restrict__ sums, const float* __restrict__ cnts,
    const float* __restrict__ fw1, const float* __restrict__ fb1,
    const float* __restrict__ fw2, const float* __restrict__ fb2,
    float* __restrict__ out) {
  __shared__ float pr[H_DIM];
  __shared__ float hid[H_DIM / 2];
  int g = blockIdx.x;
  int j = threadIdx.x;  // 64 threads
  float c = cnts[g];
  c = c > 1.f ? c : 1.f;
  float rc = 1.f / c;
  pr[j]      = sums[g * H_DIM + j] * rc;
  pr[j + 64] = sums[g * H_DIM + j + 64] * rc;
  __syncthreads();
  float a = fb1[j];
  for (int k = 0; k < H_DIM; ++k) a += pr[k] * fw1[k * (H_DIM / 2) + j];
  hid[j] = a > 0.f ? a : 0.f;
  __syncthreads();
  if (j == 0) {
    float o = fb2[0];
    for (int k = 0; k < H_DIM / 2; ++k) o += hid[k] * fw2[k];
    out[g] = o;
  }
}

// ---------------------------------------------------------------- launcher
extern "C" void kernel_launch(void* const* d_in, const int* in_sizes, int n_in,
                              void* d_out, int out_size, void* d_ws, size_t ws_size,
                              hipStream_t stream) {
  const float* x     = (const float*)d_in[0];
  const int*   ei    = (const int*)d_in[1];
  const int*   batch = (const int*)d_in[2];
  const float* W1 = (const float*)d_in[3];
  const float* b1 = (const float*)d_in[4];
  const float* g1 = (const float*)d_in[5];
  const float* t1 = (const float*)d_in[6];
  const float* W2 = (const float*)d_in[7];
  const float* b2 = (const float*)d_in[8];
  const float* g2 = (const float*)d_in[9];
  const float* t2 = (const float*)d_in[10];
  const float* W3 = (const float*)d_in[11];
  const float* b3 = (const float*)d_in[12];
  const float* g3 = (const float*)d_in[13];
  const float* t3 = (const float*)d_in[14];
  const float* fw1 = (const float*)d_in[15];
  const float* fb1 = (const float*)d_in[16];
  const float* fw2 = (const float*)d_in[17];
  const float* fb2 = (const float*)d_in[18];

  const int* src = ei;            // edge_index[0,:]
  const int* dst = ei + N_EDGES;  // edge_index[1,:]

  // Workspace layout: 2 f32 activation buffers, 1 f16 activation buffer,
  // f16 transposed weights, dis, BN stats, pool sums/counts.
  float* ws = (float*)d_ws;
  const size_t NH = (size_t)N_NODES * H_DIM;
  float*     bufA = ws;                      // xw  f32 (51.2 MB)
  float*     bufB = bufA + NH;               // agg f32 (51.2 MB)
  _Float16*  hbuf = (_Float16*)(bufB + NH);  // x/h f16 (25.6 MB)
  _Float16*  Wt   = hbuf + NH;               // 128x128 f16 (32 KB)
  float*     dis  = (float*)(Wt + F_INx * H_DIM);
  float*     stats = dis + N_NODES;                    // 512 floats
  float*     pool  = stats + 4 * H_DIM;                // G*H floats
  float*     cnts  = pool + (size_t)G_GRAPHS * H_DIM;  // G floats

  const int NHi = N_NODES * H_DIM;

  // degree -> dis = rsqrt(deg + 1)
  zero_kernel<<<(N_NODES + 255) / 256, 256, 0, stream>>>(dis, N_NODES);
  deg_accum<<<(N_EDGES + 255) / 256, 256, 0, stream>>>(dst, dis);
  deg_finalize<<<(N_NODES + 255) / 256, 256, 0, stream>>>(dis);

  // layer-1 input in f16
  f32_to_f16<<<(NHi + 255) / 256, 256, 0, stream>>>(x, hbuf, NHi);

  const float* Ws[3]  = {W1, W2, W3};
  const float* bs[3]  = {b1, b2, b3};
  const float* gs[3]  = {g1, g2, g3};
  const float* bts[3] = {t1, t2, t3};

  for (int l = 0; l < 3; ++l) {
    wt_prep<<<(F_INx * H_DIM) / 256, 256, 0, stream>>>(Ws[l], Wt);
    gemm_wmma<<<N_NODES / 16, 256, 0, stream>>>(hbuf, Wt, bs[l], bufA);
    selfloop_init<<<(NHi + 255) / 256, 256, 0, stream>>>(bufA, dis, bufB);
    scatter_edges<<<(N_EDGES * 32) / 256, 256, 0, stream>>>(bufA, src, dst, dis, bufB);
    zero_kernel<<<2, 256, 0, stream>>>(stats, 4 * H_DIM);
    bn_stats<<<(N_NODES + 255) / 256, 128, 0, stream>>>(bufB, stats);
    bn_finalize<<<1, 128, 0, stream>>>(stats, gs[l], bts[l]);
    bn_apply_relu<<<(NHi + 255) / 256, 256, 0, stream>>>(bufB, stats, hbuf);
  }

  // global mean pool + MLP head
  zero_kernel<<<(G_GRAPHS * H_DIM + 255) / 256, 256, 0, stream>>>(pool, G_GRAPHS * H_DIM);
  zero_kernel<<<(G_GRAPHS + 255) / 256, 256, 0, stream>>>(cnts, G_GRAPHS);
  pool_accum<<<N_NODES, 128, 0, stream>>>(hbuf, batch, pool, cnts);
  fc_head<<<G_GRAPHS, 64, 0, stream>>>(pool, cnts, fw1, fb1, fw2, fb2, (float*)d_out);
}
